// RelationNetwork_81965155877218
// MI455X (gfx1250) — matrix operand order repaired and checked by reading, not compiled, not run on hardware
//
#include <hip/hip_runtime.h>

// ---------------------------------------------------------------------------
// RelationNetwork on gfx1250 (MI455X, CDNA5, wave32, WMMA)
//
// Heavy op: 3x3 conv over 800 images of [128,20,20] (~94 GFLOP) -> im2col
// GEMM with v_wmma_f32_16x16x32_bf16, r1 plane resident in 320KB WGP LDS.
// Pairwise-concat 1x1 conv factorized into u[i]+v[j] (two small GEMMs).
// SiLU uses v_rcp_f32 (not IEEE divide) - precision already bf16-bounded.
// ---------------------------------------------------------------------------

typedef __attribute__((ext_vector_type(16))) __bf16 v16bf;
typedef __attribute__((ext_vector_type(8)))  float  v8f;

union Frag16 { unsigned u[8]; v16bf v; };

#define EPSBN 1e-5f
#define AS_STR 132   // LDS row stride (bf16 elems): 128 + 4 pad -> kills bank conflicts

static __device__ __forceinline__ __bf16 f2bf(float f) {
  // round-to-nearest-even f32 -> bf16
  unsigned x = __builtin_bit_cast(unsigned, f);
  unsigned r = x + 0x7fffu + ((x >> 16) & 1u);
  unsigned short h = (unsigned short)(r >> 16);
  return __builtin_bit_cast(__bf16, h);
}

static __device__ __forceinline__ float silu(float x) {
  // x * sigmoid(x) with hardware v_exp_f32 + v_rcp_f32 (avoids the IEEE
  // v_div_scale expansion; error ~1ulp of rcp, far below bf16 storage noise)
  float e = __expf(-x);
  return x * __builtin_amdgcn_rcpf(1.0f + e);
}

// One 16x16 output tile of a [16 x 128] * [128 x 128]^T GEMM.
// As: LDS A tile, row-major [16][AS_STR] bf16 (rows = M).
// W : global weights, row-major [128][128] bf16; B[k][n] = W[n][k].
// Layouts used here (A/B in contiguous k, not the ISA table order):
//   A 16x32 bf16: lane&15 = M; K = 16*(lane>>4) + 2jv (+t)
//   B 32x16 bf16: lane&15 = N; K = 16*(lane>>4) + 2jv (+t)
//   C/D f32:      lane&15 = N; M = r + 8*(lane>>4)
static __device__ __forceinline__ v8f gemm_k128(const __bf16* As, const __bf16* W,
                                                int n_base, int lane) {
  const int lcol = lane & 15, hf = lane >> 4;
  v8f acc = {0.f, 0.f, 0.f, 0.f, 0.f, 0.f, 0.f, 0.f};
  const __bf16* arow = As + lcol * AS_STR + 16 * hf;
  const __bf16* wrow = W + (n_base + lcol) * 128 + 16 * hf;
#pragma unroll
  for (int kb = 0; kb < 128; kb += 32) {
    Frag16 a, b;
#pragma unroll
    for (int jv = 0; jv < 4; ++jv) {
      a.u[jv]     = *(const unsigned*)(arow + kb + 2 * jv);
      a.u[jv + 4] = *(const unsigned*)(arow + kb + 8 + 2 * jv);
    }
#pragma unroll
    for (int jv = 0; jv < 8; ++jv)
      b.u[jv] = *(const unsigned*)(wrow + kb + 2 * jv);
    acc = __builtin_amdgcn_wmma_f32_16x16x32_bf16(false, a.v, false, b.v,
                                                  (short)0, acc, false, false);
  }
  return acc;
}

// ---------------------------------------------------------------------------
// K0: fold BatchNorm params, build bf16 weight copies (wr2 in im2col layout).
// ---------------------------------------------------------------------------
__global__ void k0_prep(
    const float* __restrict w1,  const float* __restrict g1,  const float* __restrict b1,
    const float* __restrict m1,  const float* __restrict v1,
    const float* __restrict w2,  const float* __restrict g2,  const float* __restrict b2,
    const float* __restrict m2,  const float* __restrict v2,
    const float* __restrict wr1, const float* __restrict gr1, const float* __restrict br1,
    const float* __restrict mr1, const float* __restrict vr1,
    const float* __restrict wr2, const float* __restrict gr2, const float* __restrict br2,
    const float* __restrict mr2, const float* __restrict vr2,
    const float* __restrict wr3, const float* __restrict gr3, const float* __restrict br3,
    const float* __restrict mr3, const float* __restrict vr3,
    float* __restrict s1, float* __restrict t1, float* __restrict s2, float* __restrict t2,
    float* __restrict tr1, float* __restrict tr2, float* __restrict wr3s, float* __restrict tr3,
    __bf16* __restrict w1b, __bf16* __restrict Ab, __bf16* __restrict Bb,
    __bf16* __restrict w2b, __bf16* __restrict w2im) {
  const int N_W2IM = 128 * 1152;  // wr2 im2col: [co][r*128+ci], BN scale folded
  const int N_W = 128 * 128;
  const long total = (long)N_W2IM + 3L * N_W + 128;
  for (long idx = (long)blockIdx.x * blockDim.x + threadIdx.x; idx < total;
       idx += (long)gridDim.x * blockDim.x) {
    long id = idx;
    if (id < N_W2IM) {
      int co = (int)(id / 1152), rem = (int)(id % 1152);
      int r = rem >> 7, ci = rem & 127;  // r = kh*3+kw
      float s = gr2[co] * rsqrtf(vr2[co] + EPSBN);
      w2im[id] = f2bf(s * wr2[(co * 128 + ci) * 9 + r]);
      continue;
    }
    id -= N_W2IM;
    if (id < N_W) { w1b[id] = f2bf(w1[id]); continue; }
    id -= N_W;
    if (id < N_W) {  // wr1 = [A | B], fold sr1 into both halves
      int n = (int)(id >> 7), k = (int)(id & 127);
      float s = gr1[n] * rsqrtf(vr1[n] + EPSBN);
      Ab[id] = f2bf(s * wr1[n * 256 + k]);
      Bb[id] = f2bf(s * wr1[n * 256 + 128 + k]);
      continue;
    }
    id -= N_W;
    if (id < N_W) { w2b[id] = f2bf(w2[id]); continue; }
    id -= N_W;
    int c = (int)id;  // 0..127: per-channel BN scale/shift
    float ss1 = g1[c] * rsqrtf(v1[c] + EPSBN); s1[c] = ss1; t1[c] = b1[c] - m1[c] * ss1;
    float ss2 = g2[c] * rsqrtf(v2[c] + EPSBN); s2[c] = ss2; t2[c] = b2[c] - m2[c] * ss2;
    float sr1 = gr1[c] * rsqrtf(vr1[c] + EPSBN); tr1[c] = br1[c] - mr1[c] * sr1;
    float sr2 = gr2[c] * rsqrtf(vr2[c] + EPSBN); tr2[c] = br2[c] - mr2[c] * sr2;
    float sr3 = gr3[0] * rsqrtf(vr3[0] + EPSBN);
    wr3s[c] = sr3 * wr3[c];
    if (c == 0) tr3[0] = br3[0] - mr3[0] * sr3;
  }
}

// ---------------------------------------------------------------------------
// K1: feat[m][n] = silu(s1[n]*(sum_k w1[n][k]*x[b][k][p]) + t1[n]),
// m = b*400+p. 50 blocks x 8 waves; wave w owns N-tile w.
// ---------------------------------------------------------------------------
__global__ void __launch_bounds__(256)
k1_feat(const float* __restrict x, const __bf16* __restrict w1b,
        const float* __restrict s1, const float* __restrict t1,
        float* __restrict feat, __bf16* __restrict featb) {
  __shared__ __attribute__((aligned(16))) __bf16 As[16 * AS_STR];
  const int tid = threadIdx.x, mtile = blockIdx.x;
  for (int idx = tid; idx < 16 * 128; idx += 256) {
    int mi = idx & 15, k = idx >> 4;           // consecutive tid -> consecutive p
    int m = mtile * 16 + mi;
    As[mi * AS_STR + k] = f2bf(x[((m / 400) * 128 + k) * 400 + (m % 400)]);
  }
  __syncthreads();
  const int wv = tid >> 5, lane = tid & 31;
  v8f acc = gemm_k128(As, w1b, wv * 16, lane);
  const int lcol = lane & 15, hf = lane >> 4, n = wv * 16 + lcol;
  const float sc = s1[n], tb = t1[n];
#pragma unroll
  for (int r = 0; r < 8; ++r) {
    int m = mtile * 16 + r + 8 * hf;
    float y = silu(sc * acc[r] + tb);
    feat[m * 128 + n] = y;
    featb[m * 128 + n] = f2bf(y);
  }
}

// ---------------------------------------------------------------------------
// K2: u = A'·feat, v = B'·feat (BN scale pre-folded, shift added in K3).
// gridDim.y selects u/v.
// ---------------------------------------------------------------------------
__global__ void __launch_bounds__(256)
k2_uv(const __bf16* __restrict featb, const __bf16* __restrict Ab,
      const __bf16* __restrict Bb, float* __restrict u, float* __restrict v) {
  __shared__ __attribute__((aligned(16))) __bf16 As[16 * AS_STR];
  const int tid = threadIdx.x, mtile = blockIdx.x;
  const __bf16* W = (blockIdx.y == 0) ? Ab : Bb;
  float* out = (blockIdx.y == 0) ? u : v;
  for (int idx = tid; idx < 16 * 128; idx += 256) {
    int mi = idx >> 7, k = idx & 127;
    As[mi * AS_STR + k] = featb[(mtile * 16 + mi) * 128 + k];
  }
  __syncthreads();
  const int wv = tid >> 5, lane = tid & 31;
  v8f acc = gemm_k128(As, W, wv * 16, lane);
  const int lcol = lane & 15, hf = lane >> 4, n = wv * 16 + lcol;
#pragma unroll
  for (int r = 0; r < 8; ++r)
    out[(mtile * 16 + r + 8 * hf) * 128 + n] = acc[r];
}

// ---------------------------------------------------------------------------
// K3: one block per relation row i. Builds r1[j][c] = silu(u_i+v_j+t) as bf16
// in LDS (400 x 132), runs 3x3 conv as im2col GEMM M=400,N=128,K=1152 with
// bf16 WMMA, fuses wr3 reduction + score SiLU + scores·feat aggregation.
// ---------------------------------------------------------------------------
__global__ void __launch_bounds__(256)
k3_rel(const float* __restrict u, const float* __restrict v,
       const float* __restrict tr1, const __bf16* __restrict w2im,
       const float* __restrict wr3s, const float* __restrict tr2,
       const float* __restrict tr3, const float* __restrict feat,
       float* __restrict att) {
  extern __shared__ char smem[];
  __bf16* r1s = (__bf16*)smem;                              // 400*132 bf16
  float* us = (float*)(smem + 400 * AS_STR * 2);            // 128 f32
  float* sc = us + 128;                                     // 400 f32 score accum
  const int tid = threadIdx.x, i = blockIdx.x, b = i / 400;

  if (tid < 128) us[tid] = u[i * 128 + tid] + tr1[tid];
  for (int j = tid; j < 400; j += 256) sc[j] = tr3[0];
  __syncthreads();

  const float* vb = v + b * 400 * 128;
  for (int idx = tid; idx < 400 * 128; idx += 256) {
    int j = idx >> 7, c = idx & 127;
    r1s[j * AS_STR + c] = f2bf(silu(us[c] + vb[idx]));
  }
  __syncthreads();

  const int wv = tid >> 5, lane = tid & 31, lcol = lane & 15, hf = lane >> 4;
  const int co = wv * 16 + lcol;                 // wave wv owns N-tile wv
  const float w3 = wr3s[co], tb2 = tr2[co];
  const __bf16* wrow = w2im + co * 1152 + 16 * hf;

  for (int mg = 0; mg < 5; ++mg) {               // 25 M-tiles in groups of 5
    v8f acc[5];
#pragma unroll
    for (int t = 0; t < 5; ++t)
#pragma unroll
      for (int r = 0; r < 8; ++r) acc[t][r] = 0.f;

    for (int kc = 0; kc < 36; ++kc) {            // K = 9 taps * 4 ci-chunks of 32
      const int r9 = kc >> 2, cblk = kc & 3;
      const int dh = r9 / 3 - 1, dw = r9 % 3 - 1;
      Frag16 bf;
#pragma unroll
      for (int jv = 0; jv < 8; ++jv)
        bf.u[jv] = *(const unsigned*)(wrow + kc * 32 + 2 * jv);
#pragma unroll
      for (int t = 0; t < 5; ++t) {
        const int j = (mg * 5 + t) * 16 + lcol;
        const int hh = j / 20 + dh, ww = j % 20 + dw;
        Frag16 a;
        if ((unsigned)hh < 20u && (unsigned)ww < 20u) {
          const __bf16* ar = r1s + (hh * 20 + ww) * AS_STR + cblk * 32 + 16 * hf;
#pragma unroll
          for (int jv = 0; jv < 4; ++jv) {
            a.u[jv]     = *(const unsigned*)(ar + 2 * jv);
            a.u[jv + 4] = *(const unsigned*)(ar + 8 + 2 * jv);
          }
        } else {
#pragma unroll
          for (int jv = 0; jv < 8; ++jv) a.u[jv] = 0u;     // zero padding
        }
        acc[t] = __builtin_amdgcn_wmma_f32_16x16x32_bf16(false, a.v, false, bf.v,
                                                         (short)0, acc[t], false, false);
      }
    }
    // fused BN+SiLU then wr3 dot: reduce over co (across the 16 lanes)
#pragma unroll
    for (int t = 0; t < 5; ++t) {
      const int mtile = mg * 5 + t;
#pragma unroll
      for (int rr = 0; rr < 8; ++rr) {
        float p = w3 * silu(acc[t][rr] + tb2);
        p += __shfl_xor(p, 1); p += __shfl_xor(p, 2);
        p += __shfl_xor(p, 4); p += __shfl_xor(p, 8);
        if (lcol == 0) atomicAdd(&sc[mtile * 16 + rr + 8 * hf], p);
      }
    }
  }
  __syncthreads();
  for (int j = tid; j < 400; j += 256) sc[j] = silu(sc[j]);  // scores[i][j]
  __syncthreads();
  if (tid < 128) {                               // att[i][c] = sum_j sc[j]*feat[b,j,c]
    float a = 0.f;
    const float* fb = feat + b * 400 * 128 + tid;
    for (int j = 0; j < 400; ++j) a = fmaf(sc[j], fb[j * 128], a);
    att[i * 128 + tid] = a;
  }
}

// ---------------------------------------------------------------------------
// K4: out = silu(s2*(w2·att) + t2) + x
// ---------------------------------------------------------------------------
__global__ void __launch_bounds__(256)
k4_out(const float* __restrict att, const __bf16* __restrict w2b,
       const float* __restrict s2, const float* __restrict t2,
       const float* __restrict x, float* __restrict out) {
  __shared__ __attribute__((aligned(16))) __bf16 As[16 * AS_STR];
  const int tid = threadIdx.x, mtile = blockIdx.x;
  for (int idx = tid; idx < 16 * 128; idx += 256) {
    int mi = idx >> 7, k = idx & 127;
    As[mi * AS_STR + k] = f2bf(att[(mtile * 16 + mi) * 128 + k]);
  }
  __syncthreads();
  const int wv = tid >> 5, lane = tid & 31;
  v8f acc = gemm_k128(As, w2b, wv * 16, lane);
  const int lcol = lane & 15, hf = lane >> 4, n = wv * 16 + lcol;
  const float scn = s2[n], tbn = t2[n];
#pragma unroll
  for (int r = 0; r < 8; ++r) {
    int m = mtile * 16 + r + 8 * hf;
    int b = m / 400, p = m % 400;
    out[(b * 128 + n) * 400 + p] = silu(scn * acc[r] + tbn) + x[(b * 128 + n) * 400 + p];
  }
}

// ---------------------------------------------------------------------------
extern "C" void kernel_launch(void* const* d_in, const int* in_sizes, int n_in,
                              void* d_out, int out_size, void* d_ws, size_t ws_size,
                              hipStream_t stream) {
  (void)in_sizes; (void)n_in; (void)out_size; (void)ws_size;
  const float* x   = (const float*)d_in[0];
  const float* w1  = (const float*)d_in[1];
  const float* g1  = (const float*)d_in[2];
  const float* b1  = (const float*)d_in[3];
  const float* m1  = (const float*)d_in[4];
  const float* v1  = (const float*)d_in[5];
  const float* w2  = (const float*)d_in[6];
  const float* g2  = (const float*)d_in[7];
  const float* b2  = (const float*)d_in[8];
  const float* m2  = (const float*)d_in[9];
  const float* v2  = (const float*)d_in[10];
  const float* wr1 = (const float*)d_in[11];
  const float* gr1 = (const float*)d_in[12];
  const float* br1 = (const float*)d_in[13];
  const float* mr1 = (const float*)d_in[14];
  const float* vr1 = (const float*)d_in[15];
  const float* wr2 = (const float*)d_in[16];
  const float* gr2 = (const float*)d_in[17];
  const float* br2 = (const float*)d_in[18];
  const float* mr2 = (const float*)d_in[19];
  const float* vr2 = (const float*)d_in[20];
  const float* wr3 = (const float*)d_in[21];
  const float* gr3 = (const float*)d_in[22];
  const float* br3 = (const float*)d_in[23];
  const float* mr3 = (const float*)d_in[24];
  const float* vr3 = (const float*)d_in[25];
  float* outp = (float*)d_out;

  // workspace carve: f32 region then bf16 region
  float* wsf  = (float*)d_ws;
  float* s1   = wsf + 0;       float* t1  = wsf + 128;
  float* s2   = wsf + 256;     float* t2  = wsf + 384;
  float* tr1  = wsf + 512;     float* tr2 = wsf + 640;
  float* wr3s = wsf + 768;     float* tr3 = wsf + 896;    // 128-float pad slot
  float* feat = wsf + 1024;                                // 800*128
  float* u    = feat + 800 * 128;
  float* v    = u    + 800 * 128;
  float* att  = v    + 800 * 128;
  char* bfbase = (char*)d_ws + (size_t)(1024 + 4 * 800 * 128) * sizeof(float);
  __bf16* w1b   = (__bf16*)bfbase;            // 128*128
  __bf16* Ab    = w1b  + 128 * 128;
  __bf16* Bb    = Ab   + 128 * 128;
  __bf16* w2b   = Bb   + 128 * 128;
  __bf16* w2im  = w2b  + 128 * 128;           // 128*1152
  __bf16* featb = w2im + 128 * 1152;          // 800*128

  k0_prep<<<192, 256, 0, stream>>>(w1, g1, b1, m1, v1, w2, g2, b2, m2, v2,
                                   wr1, gr1, br1, mr1, vr1, wr2, gr2, br2, mr2, vr2,
                                   wr3, gr3, br3, mr3, vr3,
                                   s1, t1, s2, t2, tr1, tr2, wr3s, tr3,
                                   w1b, Ab, Bb, w2b, w2im);
  k1_feat<<<50, 256, 0, stream>>>(x, w1b, s1, t1, feat, featb);
  k2_uv<<<dim3(50, 2), 256, 0, stream>>>(featb, Ab, Bb, u, v);
  const size_t sm3 = (size_t)(400 * AS_STR * 2) + 128 * 4 + 400 * 4;  // 107712 B
  k3_rel<<<800, 256, sm3, stream>>>(u, v, tr1, w2im, wr3s, tr2, tr3, feat, att);
  k4_out<<<50, 256, 0, stream>>>(att, w2b, s2, t2, x, outp);
}
